// PseudoQuantizedQwen3_5MoeExperts_72241349919184
// MI455X (gfx1250) — matrix operand order repaired and checked by reading, not compiled, not run on hardware
//
#include <hip/hip_runtime.h>
#include <hip/hip_bf16.h>
#include <math.h>

#define T_TOKENS 4096
#define HIDDEN   2048
#define INTER    768
#define NEXP     8
#define TOPK     2
#define NROT     4
#define GS       128
#define MAX_TILES 520           // sum_e ceil(count_e/16) <= 8192/16 + 8
#define PAD_ROWS (MAX_TILES*16) // 8320

// meta[] layout (ints): [0..7]=counts, [8..15]=cursor, [16..23]=expert_off,
// [32..551]=tile_expert, [560..560+PAD_ROWS)=entry_token,
// [560+PAD_ROWS .. +8192)=pos_of
#define META_TILE  32
#define META_ETOK  560
#define META_POS   (560 + PAD_ROWS)

typedef __attribute__((ext_vector_type(16))) __bf16 v16bf;
typedef __attribute__((ext_vector_type(8)))  float  v8f;

union FragU { v16bf v; uint4 q[2]; };

// A-matrix 16x32 bf16 fragment: lane(sub,half) holds
// K = half*16 .. half*16+15 contiguous (same order as B).
__device__ __forceinline__ v16bf load_fragA(const __hip_bfloat16* row, int k0, int half) {
  FragU u;
  u.q[0] = *reinterpret_cast<const uint4*>(row + k0 + half * 16);
  u.q[1] = *reinterpret_cast<const uint4*>(row + k0 + half * 16 + 8);
  return u.v;
}
// B-matrix 32x16 bf16 fragment: lane(sub,half) holds K = half*16 .. half*16+15
// contiguous (column n=sub of B == weight row o, contiguous in k).
__device__ __forceinline__ v16bf load_fragB(const __hip_bfloat16* row, int k0, int half) {
  const uint4* p = reinterpret_cast<const uint4*>(row + k0 + half * 16);
  FragU u;
  u.q[0] = p[0];
  u.q[1] = p[1];
  return u.v;
}

// ---------------- Stage 1: weight pseudo-quantization -> bf16 ----------------
__global__ __launch_bounds__(128) void pseudo_quant_kernel(
    const float* __restrict__ W, const int* __restrict__ pairs,
    const float* __restrict__ angles, const float* __restrict__ chs,
    __hip_bfloat16* __restrict__ out, int In)
{
  const int row = blockIdx.y;
  const int g   = blockIdx.x;
  const int tid = threadIdx.x;
  const int halfIn = In >> 1;
  const size_t base = (size_t)row * In + (size_t)g * GS;

  __shared__ float w[GS];
  __shared__ float red[GS];
  __shared__ float s_max, s_min;

  const float cs = chs[g * GS + tid];
  w[tid] = W[base + tid] * cs;
  __syncthreads();

  // forward rotations r = 0..3 (64 disjoint pairs per pass)
  for (int r = 0; r < NROT; ++r) {
    if (tid < 64) {
      const int pb = r * In + g * GS + 2 * tid;
      const int il = pairs[pb], jl = pairs[pb + 1];
      const float a = angles[r * halfIn + g * 64 + tid];
      const float c = cosf(a), s = sinf(a);
      const float wi = w[il], wj = w[jl];
      w[il] = c * wi - s * wj;
      w[jl] = s * wi + c * wj;
    }
    __syncthreads();
  }

  // group max / min
  red[tid] = w[tid]; __syncthreads();
  for (int st = 64; st > 0; st >>= 1) { if (tid < st) red[tid] = fmaxf(red[tid], red[tid + st]); __syncthreads(); }
  if (tid == 0) s_max = red[0];
  __syncthreads();
  red[tid] = w[tid]; __syncthreads();
  for (int st = 64; st > 0; st >>= 1) { if (tid < st) red[tid] = fminf(red[tid], red[tid + st]); __syncthreads(); }
  if (tid == 0) s_min = red[0];
  __syncthreads();

  // 4-bit fake quant (RNE like jnp.round)
  const float scale = fmaxf(s_max - s_min, 1e-5f) * (1.0f / 15.0f);
  const float zero  = rintf(-s_min / scale);
  const float q = fminf(fmaxf(rintf(w[tid] / scale) + zero, 0.0f), 15.0f);
  __syncthreads();
  w[tid] = (q - zero) * scale;
  __syncthreads();

  // inverse rotations: reversed order, negated angle
  for (int r = NROT - 1; r >= 0; --r) {
    if (tid < 64) {
      const int pb = r * In + g * GS + 2 * tid;
      const int il = pairs[pb], jl = pairs[pb + 1];
      const float a = angles[r * halfIn + g * 64 + tid];
      const float c = cosf(a), s = sinf(a);
      const float wi = w[il], wj = w[jl];
      w[il] =  c * wi + s * wj;
      w[jl] = -s * wi + c * wj;
    }
    __syncthreads();
  }
  out[base + tid] = __float2bfloat16(w[tid] / cs);
}

__global__ void f32_to_bf16_kernel(const float* __restrict__ in,
                                   __hip_bfloat16* __restrict__ out, int n) {
  int i = blockIdx.x * blockDim.x + threadIdx.x;
  if (i < n) out[i] = __float2bfloat16(in[i]);
}

// ---------------- Stage 2: device-side routing ----------------
__global__ void route_init_kernel(int* meta) {
  int i = threadIdx.x;
  if (i < 8) { meta[i] = 0; meta[8 + i] = 0; }
  for (int t = i; t < MAX_TILES; t += blockDim.x) meta[META_TILE + t] = -1;
}

__global__ void route_count_kernel(const int* __restrict__ idx, int* meta) {
  int i = blockIdx.x * blockDim.x + threadIdx.x;
  if (i < T_TOKENS * TOPK) atomicAdd(&meta[idx[i]], 1);
}

__global__ void route_offsets_kernel(int* meta) {
  if (threadIdx.x == 0 && blockIdx.x == 0) {
    int off = 0, tile = 0;
    for (int e = 0; e < NEXP; ++e) {
      const int c  = meta[e];
      const int pc = (c + 15) & ~15;
      meta[16 + e] = off;
      for (int j = 0; j < (pc >> 4); ++j) meta[META_TILE + tile++] = e;
      for (int j = c; j < pc; ++j) meta[META_ETOK + off + j] = 0; // pad token
      off += pc;
    }
  }
}

__global__ void route_scatter_kernel(const int* __restrict__ idx, int* meta) {
  int i = blockIdx.x * blockDim.x + threadIdx.x;
  if (i >= T_TOKENS * TOPK) return;
  const int e   = idx[i];
  const int pos = atomicAdd(&meta[8 + e], 1);
  const int p   = meta[16 + e] + pos;
  meta[META_ETOK + p] = i >> 1;  // token for this routed slot
  meta[META_POS + i]  = p;       // where (t,k) landed
}

// ---------------- Stage 3: gate/up GEMM + SiLU (bf16 WMMA) ----------------
// One wave per (16-row tile, 32 inter columns): two (gate,up) column pairs.
// Explicit double-buffering: next k-step's 5 fragment loads are issued before
// this k-step's 4 WMMAs so waits are partial (loads overlap the matrix pipe).
__global__ __launch_bounds__(32) void gemm1_kernel(
    const __hip_bfloat16* __restrict__ hid,
    const __hip_bfloat16* __restrict__ gup,
    const int* __restrict__ meta,
    __hip_bfloat16* __restrict__ inter)
{
  const int mt = blockIdx.y;
  const int e  = meta[META_TILE + mt];
  if (e < 0) return;
  const int lane = threadIdx.x;
  const int sub = lane & 15, half = lane >> 4;

  const int tok = meta[META_ETOK + mt * 16 + sub];
  const __hip_bfloat16* aRow = hid + (size_t)tok * HIDDEN;
  const int og = blockIdx.x * 32 + sub;      // first gate row; second = og+16
  const __hip_bfloat16* bg0 = gup + ((size_t)e * (2 * INTER) + og) * HIDDEN;
  const __hip_bfloat16* bg1 = bg0 + (size_t)16 * HIDDEN;
  const __hip_bfloat16* bu0 = bg0 + (size_t)INTER * HIDDEN;
  const __hip_bfloat16* bu1 = bu0 + (size_t)16 * HIDDEN;

  v8f cg0 = {}; v8f cg1 = {}; v8f cu0 = {}; v8f cu1 = {};

  v16bf a   = load_fragA(aRow, 0, half);
  v16bf fg0 = load_fragB(bg0, 0, half);
  v16bf fg1 = load_fragB(bg1, 0, half);
  v16bf fu0 = load_fragB(bu0, 0, half);
  v16bf fu1 = load_fragB(bu1, 0, half);

  for (int k0 = 32; k0 < HIDDEN; k0 += 32) {
    v16bf an   = load_fragA(aRow, k0, half);
    v16bf ng0  = load_fragB(bg0, k0, half);
    v16bf ng1  = load_fragB(bg1, k0, half);
    v16bf nu0  = load_fragB(bu0, k0, half);
    v16bf nu1  = load_fragB(bu1, k0, half);
    cg0 = __builtin_amdgcn_wmma_f32_16x16x32_bf16(false, a, false, fg0, (short)0, cg0, false, false);
    cg1 = __builtin_amdgcn_wmma_f32_16x16x32_bf16(false, a, false, fg1, (short)0, cg1, false, false);
    cu0 = __builtin_amdgcn_wmma_f32_16x16x32_bf16(false, a, false, fu0, (short)0, cu0, false, false);
    cu1 = __builtin_amdgcn_wmma_f32_16x16x32_bf16(false, a, false, fu1, (short)0, cu1, false, false);
    a = an; fg0 = ng0; fg1 = ng1; fu0 = nu0; fu1 = nu1;
  }
  cg0 = __builtin_amdgcn_wmma_f32_16x16x32_bf16(false, a, false, fg0, (short)0, cg0, false, false);
  cg1 = __builtin_amdgcn_wmma_f32_16x16x32_bf16(false, a, false, fg1, (short)0, cg1, false, false);
  cu0 = __builtin_amdgcn_wmma_f32_16x16x32_bf16(false, a, false, fu0, (short)0, cu0, false, false);
  cu1 = __builtin_amdgcn_wmma_f32_16x16x32_bf16(false, a, false, fu1, (short)0, cu1, false, false);

  // C/D layout: lane -> N = sub, M = j + half*8
  const size_t p0 = (size_t)mt * 16;
  const int col = blockIdx.x * 32 + sub;
#pragma unroll
  for (int j = 0; j < 8; ++j) {
    const size_t r = (p0 + j + half * 8) * INTER;
    const float g0 = cg0[j], u0 = cu0[j];
    const float g1 = cg1[j], u1 = cu1[j];
    inter[r + col]      = __float2bfloat16((g0 / (1.0f + __expf(-g0))) * u0);
    inter[r + col + 16] = __float2bfloat16((g1 / (1.0f + __expf(-g1))) * u1);
  }
}

// ---------------- Stage 4: down GEMM (bf16 WMMA, 4 N-tiles/wave) ----------------
__global__ __launch_bounds__(32) void gemm2_kernel(
    const __hip_bfloat16* __restrict__ inter,
    const __hip_bfloat16* __restrict__ dwn,
    const int* __restrict__ meta,
    __hip_bfloat16* __restrict__ y)
{
  const int mt = blockIdx.y;
  const int e  = meta[META_TILE + mt];
  if (e < 0) return;
  const int lane = threadIdx.x;
  const int sub = lane & 15, half = lane >> 4;

  const __hip_bfloat16* aRow = inter + ((size_t)mt * 16 + sub) * INTER;
  const int h0 = blockIdx.x * 64;
  const __hip_bfloat16* b0 = dwn + ((size_t)e * HIDDEN + h0 + sub) * INTER;
  const __hip_bfloat16* b1 = b0 + (size_t)16 * INTER;
  const __hip_bfloat16* b2 = b0 + (size_t)32 * INTER;
  const __hip_bfloat16* b3 = b0 + (size_t)48 * INTER;

  v8f c0 = {}; v8f c1 = {}; v8f c2 = {}; v8f c3 = {};

  v16bf a  = load_fragA(aRow, 0, half);
  v16bf f0 = load_fragB(b0, 0, half);
  v16bf f1 = load_fragB(b1, 0, half);
  v16bf f2 = load_fragB(b2, 0, half);
  v16bf f3 = load_fragB(b3, 0, half);

  for (int k0 = 32; k0 < INTER; k0 += 32) {
    v16bf an = load_fragA(aRow, k0, half);
    v16bf n0 = load_fragB(b0, k0, half);
    v16bf n1 = load_fragB(b1, k0, half);
    v16bf n2 = load_fragB(b2, k0, half);
    v16bf n3 = load_fragB(b3, k0, half);
    c0 = __builtin_amdgcn_wmma_f32_16x16x32_bf16(false, a, false, f0, (short)0, c0, false, false);
    c1 = __builtin_amdgcn_wmma_f32_16x16x32_bf16(false, a, false, f1, (short)0, c1, false, false);
    c2 = __builtin_amdgcn_wmma_f32_16x16x32_bf16(false, a, false, f2, (short)0, c2, false, false);
    c3 = __builtin_amdgcn_wmma_f32_16x16x32_bf16(false, a, false, f3, (short)0, c3, false, false);
    a = an; f0 = n0; f1 = n1; f2 = n2; f3 = n3;
  }
  c0 = __builtin_amdgcn_wmma_f32_16x16x32_bf16(false, a, false, f0, (short)0, c0, false, false);
  c1 = __builtin_amdgcn_wmma_f32_16x16x32_bf16(false, a, false, f1, (short)0, c1, false, false);
  c2 = __builtin_amdgcn_wmma_f32_16x16x32_bf16(false, a, false, f2, (short)0, c2, false, false);
  c3 = __builtin_amdgcn_wmma_f32_16x16x32_bf16(false, a, false, f3, (short)0, c3, false, false);

  const size_t p0 = (size_t)mt * 16;
#pragma unroll
  for (int j = 0; j < 8; ++j) {
    const size_t r = (p0 + j + half * 8) * HIDDEN;
    y[r + h0 + sub]      = __float2bfloat16(c0[j]);
    y[r + h0 + 16 + sub] = __float2bfloat16(c1[j]);
    y[r + h0 + 32 + sub] = __float2bfloat16(c2[j]);
    y[r + h0 + 48 + sub] = __float2bfloat16(c3[j]);
  }
}

// ---------------- Stage 5: deterministic combine (gather, no atomics) ----------------
__global__ void combine_kernel(const float* __restrict__ topw,
                               const int* __restrict__ meta,
                               const __hip_bfloat16* __restrict__ y,
                               float* __restrict__ out)
{
  int i = blockIdx.x * blockDim.x + threadIdx.x;
  if (i >= T_TOKENS * HIDDEN) return;
  const int t = i >> 11;          // / HIDDEN
  const int h = i & (HIDDEN - 1);
  const int* pos = meta + META_POS;
  const int p0 = pos[2 * t], p1 = pos[2 * t + 1];
  const float w0 = topw[2 * t], w1 = topw[2 * t + 1];
  out[i] = w0 * __bfloat162float(y[(size_t)p0 * HIDDEN + h]) +
           w1 * __bfloat162float(y[(size_t)p1 * HIDDEN + h]);
}

extern "C" void kernel_launch(void* const* d_in, const int* in_sizes, int n_in,
                              void* d_out, int out_size, void* d_ws, size_t ws_size,
                              hipStream_t stream)
{
  const float* hidden = (const float*)d_in[0];
  const int*   topidx = (const int*)  d_in[1];
  const float* topw   = (const float*)d_in[2];
  const float* gupW   = (const float*)d_in[3];
  const float* dwnW   = (const float*)d_in[4];
  const int*   gpairs = (const int*)  d_in[5];
  const float* gang   = (const float*)d_in[6];
  const int*   dpairs = (const int*)  d_in[7];
  const float* dang   = (const float*)d_in[8];
  const float* gchs   = (const float*)d_in[9];
  const float* dchs   = (const float*)d_in[10];
  float* out = (float*)d_out;

  // workspace carve-up (~139 MB total)
  char* w = (char*)d_ws;
  __hip_bfloat16* gup_q = (__hip_bfloat16*)w; w += (size_t)NEXP * 2 * INTER * HIDDEN * 2;  // 50.3 MB
  __hip_bfloat16* dwn_q = (__hip_bfloat16*)w; w += (size_t)NEXP * HIDDEN * INTER * 2;      // 25.2 MB
  __hip_bfloat16* hid_q = (__hip_bfloat16*)w; w += (size_t)T_TOKENS * HIDDEN * 2;          // 16.8 MB
  __hip_bfloat16* interB= (__hip_bfloat16*)w; w += (size_t)PAD_ROWS * INTER * 2;           // 12.8 MB
  __hip_bfloat16* ybuf  = (__hip_bfloat16*)w; w += (size_t)PAD_ROWS * HIDDEN * 2;          // 34.1 MB
  int* meta = (int*)w;                                                                     // ~68 KB

  route_init_kernel<<<1, 256, 0, stream>>>(meta);

  pseudo_quant_kernel<<<dim3(HIDDEN / GS, NEXP * 2 * INTER), 128, 0, stream>>>(
      gupW, gpairs, gang, gchs, gup_q, HIDDEN);
  pseudo_quant_kernel<<<dim3(INTER / GS, NEXP * HIDDEN), 128, 0, stream>>>(
      dwnW, dpairs, dang, dchs, dwn_q, INTER);

  f32_to_bf16_kernel<<<(T_TOKENS * HIDDEN + 255) / 256, 256, 0, stream>>>(
      hidden, hid_q, T_TOKENS * HIDDEN);

  route_count_kernel<<<(T_TOKENS * TOPK + 255) / 256, 256, 0, stream>>>(topidx, meta);
  route_offsets_kernel<<<1, 32, 0, stream>>>(meta);
  route_scatter_kernel<<<(T_TOKENS * TOPK + 255) / 256, 256, 0, stream>>>(topidx, meta);

  gemm1_kernel<<<dim3(INTER / 32, MAX_TILES), 32, 0, stream>>>(hid_q, gup_q, meta, interB);
  gemm2_kernel<<<dim3(HIDDEN / 64, MAX_TILES), 32, 0, stream>>>(interB, dwn_q, meta, ybuf);

  combine_kernel<<<(T_TOKENS * HIDDEN + 255) / 256, 256, 0, stream>>>(topw, meta, ybuf, out);
}
